// STM_29214367547768
// MI455X (gfx1250) — matrix operand order repaired and checked by reading, not compiled, not run on hardware
//
#include <hip/hip_runtime.h>
#include <hip/hip_bf16.h>

#define T_    32
#define B_    128
#define IN_   96
#define SLOT_ 96
#define NS_   8
#define H_    4
#define KEY_  32
#define VAL_  24
#define REL_  96
#define OUTA_ 64
#define EPS_  1e-5f

typedef __bf16 bf16;
typedef __attribute__((ext_vector_type(16))) __bf16 v16bf;
typedef __attribute__((ext_vector_type(8)))  __bf16 v8bf;
typedef __attribute__((ext_vector_type(8)))  float  v8f;

// ---------------------------------------------------------------------------
// Generic bf16 WMMA GEMM: C[M,N] = A[M,K] @ B[K,N] + bias[N]
// A,B f32 in global, converted to bf16 while staging into LDS.
// Block = 256 threads (8 waves); block tile 64(M) x 32(N); K-slice 64
// (+32 tail). A tile row-major 64x64; B tile stored column-major 32(N)x64(K)
// so every lane's fragment is two contiguous 16-byte LDS loads.
// Requires M%64==0, N%32==0, K%32==0.
// ---------------------------------------------------------------------------
__global__ void gemm_bf16_wmma(const float* __restrict__ A,
                               const float* __restrict__ Bm,
                               const float* __restrict__ bias,
                               float* __restrict__ Cf,
                               int M, int N, int K) {
  __shared__ __align__(16) bf16 lA[64 * 64];   // [m][k]
  __shared__ __align__(16) bf16 lB[32 * 64];   // [n][k]  (column-major vs K)
  const int tid  = threadIdx.x;
  const int nbn  = N >> 5;
  const int bm   = blockIdx.x / nbn;
  const int bn   = blockIdx.x % nbn;
  const int m0   = bm << 6;
  const int n0   = bn << 5;
  const int wave = tid >> 5;
  const int lane = tid & 31;
  const int mi   = wave & 3;          // 4 M-subtiles
  const int ni   = wave >> 2;         // 2 N-subtiles
  const int half = lane >> 4;
  const int r    = lane & 15;

  v8f acc;
#pragma unroll
  for (int i = 0; i < 8; ++i) acc[i] = 0.0f;

  int k0 = 0;
  // ---- main loop: 64-wide K slices, 2 WMMA per wave per iteration ----
  for (; k0 + 64 <= K; k0 += 64) {
    __syncthreads();
    {
      // A: 64x64 = 4096 elems, 16 per thread (4x float4), contiguous store
      const int row = tid >> 2, col0 = (tid & 3) << 4;
      const float4* gA = (const float4*)&A[(long)(m0 + row) * K + k0 + col0];
      if (k0 + 128 <= K)
        __builtin_prefetch(&A[(long)(m0 + row) * K + k0 + 64 + col0], 0, 1);
      bf16 tmp[16];
#pragma unroll
      for (int i = 0; i < 4; ++i) {
        const float4 f = gA[i];
        tmp[i * 4 + 0] = (bf16)f.x; tmp[i * 4 + 1] = (bf16)f.y;
        tmp[i * 4 + 2] = (bf16)f.z; tmp[i * 4 + 3] = (bf16)f.w;
      }
      *(v8bf*)&lA[row * 64 + col0]     = *(v8bf*)&tmp[0];
      *(v8bf*)&lA[row * 64 + col0 + 8] = *(v8bf*)&tmp[8];
    }
    {
      // B: 64(k)x32(n) = 2048 elems, 8 per thread (2x float4), scatter to [n][k]
      const int krow = tid >> 2, nc0 = (tid & 3) << 3;
      const float4* gB = (const float4*)&Bm[(long)(k0 + krow) * N + n0 + nc0];
      if (k0 + 128 <= K)
        __builtin_prefetch(&Bm[(long)(k0 + 64 + krow) * N + n0 + nc0], 0, 1);
      const float4 f0 = gB[0], f1 = gB[1];
      lB[(nc0 + 0) * 64 + krow] = (bf16)f0.x;
      lB[(nc0 + 1) * 64 + krow] = (bf16)f0.y;
      lB[(nc0 + 2) * 64 + krow] = (bf16)f0.z;
      lB[(nc0 + 3) * 64 + krow] = (bf16)f0.w;
      lB[(nc0 + 4) * 64 + krow] = (bf16)f1.x;
      lB[(nc0 + 5) * 64 + krow] = (bf16)f1.y;
      lB[(nc0 + 6) * 64 + krow] = (bf16)f1.z;
      lB[(nc0 + 7) * 64 + krow] = (bf16)f1.w;
    }
    __syncthreads();
#pragma unroll
    for (int kb = 0; kb < 2; ++kb) {
      // A fragment: lane holds M=mi*16+r; elems e<8 -> K=kb*32+half*8+e,
      // e>=8 -> K=kb*32+16+half*8+(e-8). Two aligned 16B LDS loads.
      const int abase = (mi * 16 + r) * 64 + kb * 32 + (half << 3);
      const v8bf a0 = *(const v8bf*)&lA[abase];
      const v8bf a1 = *(const v8bf*)&lA[abase + 16];
      const v16bf af = __builtin_shufflevector(a0, a1, 0, 1, 2, 3, 4, 5, 6, 7,
                                               8, 9, 10, 11, 12, 13, 14, 15);
      // B fragment: lane holds N=ni*16+r; elems e<8 -> K=kb*32+half*8+e,
      // e>=8 -> K=kb*32+16+half*8+(e-8).
      const int bbase = (ni * 16 + r) * 64 + kb * 32 + (half << 3);
      const v8bf b0 = *(const v8bf*)&lB[bbase];
      const v8bf b1 = *(const v8bf*)&lB[bbase + 16];
      const v16bf bfv = __builtin_shufflevector(b0, b1, 0, 1, 2, 3, 4, 5, 6, 7,
                                                8, 9, 10, 11, 12, 13, 14, 15);
      acc = __builtin_amdgcn_wmma_f32_16x16x32_bf16(
          false, af, false, bfv, (short)0, acc, false, false);
    }
  }
  // ---- 32-wide K tail (for K % 64 == 32, e.g. K=96) ----
  if (k0 < K) {
    __syncthreads();
    {
      const int row = tid >> 2, col0 = (tid & 3) << 3;   // 8 elems/thread
      const float4* gA = (const float4*)&A[(long)(m0 + row) * K + k0 + col0];
      const float4 f0 = gA[0], f1 = gA[1];
      bf16 tmp[8];
      tmp[0] = (bf16)f0.x; tmp[1] = (bf16)f0.y; tmp[2] = (bf16)f0.z; tmp[3] = (bf16)f0.w;
      tmp[4] = (bf16)f1.x; tmp[5] = (bf16)f1.y; tmp[6] = (bf16)f1.z; tmp[7] = (bf16)f1.w;
      *(v8bf*)&lA[row * 64 + col0] = *(v8bf*)&tmp[0];
    }
    {
      const int krow = tid >> 3, nc0 = (tid & 7) << 2;   // 4 elems/thread
      const float4 f = *(const float4*)&Bm[(long)(k0 + krow) * N + n0 + nc0];
      lB[(nc0 + 0) * 64 + krow] = (bf16)f.x;
      lB[(nc0 + 1) * 64 + krow] = (bf16)f.y;
      lB[(nc0 + 2) * 64 + krow] = (bf16)f.z;
      lB[(nc0 + 3) * 64 + krow] = (bf16)f.w;
    }
    __syncthreads();
    const int abase = (mi * 16 + r) * 64 + (half << 3);
    const v8bf a0 = *(const v8bf*)&lA[abase];
    const v8bf a1 = *(const v8bf*)&lA[abase + 16];
    const v16bf af = __builtin_shufflevector(a0, a1, 0, 1, 2, 3, 4, 5, 6, 7,
                                             8, 9, 10, 11, 12, 13, 14, 15);
    const int bbase = (ni * 16 + r) * 64 + (half << 3);
    const v8bf b0 = *(const v8bf*)&lB[bbase];
    const v8bf b1 = *(const v8bf*)&lB[bbase + 16];
    const v16bf bfv = __builtin_shufflevector(b0, b1, 0, 1, 2, 3, 4, 5, 6, 7,
                                              8, 9, 10, 11, 12, 13, 14, 15);
    acc = __builtin_amdgcn_wmma_f32_16x16x32_bf16(
        false, af, false, bfv, (short)0, acc, false, false);
  }

  // C/D layout: VGPR v holds M = v + 8*half, N = lane%16
#pragma unroll
  for (int v = 0; v < 8; ++v) {
    const int row = m0 + mi * 16 + v + (half << 3);
    const int col = n0 + ni * 16 + r;
    Cf[(long)row * N + col] = acc[v] + bias[col];
  }
}

// ---------------------------------------------------------------------------
// input_step (T,B,96) -> x_bt (B*T,96)
// ---------------------------------------------------------------------------
__global__ void transpose_in(const float* __restrict__ in,
                             float* __restrict__ out) {
  const int total = T_ * B_ * IN_;
  int idx = blockIdx.x * blockDim.x + threadIdx.x;
  if (idx >= total) return;
  const int f = idx % IN_;
  const int bt = idx / IN_;
  const int b = bt / T_;
  const int t = bt % T_;
  out[idx] = in[(t * B_ + b) * IN_ + f];
}

// ---------------------------------------------------------------------------
// Attention: softmax(q.k/sqrt(32)) over t, keep top-3, Mi_new = scores @ v.
// One thread per (b,h,n).
// ---------------------------------------------------------------------------
__global__ void attention_topk(const float* __restrict__ q,
                               const float* __restrict__ kmat,
                               const float* __restrict__ vmat,
                               float* __restrict__ Mi_new) {
  int idx = blockIdx.x * blockDim.x + threadIdx.x;
  if (idx >= B_ * H_ * NS_) return;
  const int n = idx % NS_;
  const int h = (idx / NS_) % H_;
  const int b = idx / (NS_ * H_);

  const float* qp = q + (b * NS_ + n) * (H_ * KEY_) + h * KEY_;
  float sc[T_];
  float mx = -1e30f;
  for (int t = 0; t < T_; ++t) {
    const float* kp = kmat + (b * T_ + t) * (H_ * KEY_) + h * KEY_;
    float d = 0.0f;
#pragma unroll
    for (int k = 0; k < KEY_; ++k) d += qp[k] * kp[k];
    d *= 0.17677669529663687f;  // 1/sqrt(32)
    sc[t] = d;
    mx = fmaxf(mx, d);
  }
  float se = 0.0f;
  for (int t = 0; t < T_; ++t) { sc[t] = expf(sc[t] - mx); se += sc[t]; }
  const float inv = 1.0f / se;
  for (int t = 0; t < T_; ++t) sc[t] *= inv;

  int i1 = 0, i2 = -1, i3 = -1;
  float v1 = -1e30f, v2 = -1e30f, v3 = -1e30f;
  for (int t = 0; t < T_; ++t) {
    const float v = sc[t];
    if (v > v1)      { v3 = v2; i3 = i2; v2 = v1; i2 = i1; v1 = v; i1 = t; }
    else if (v > v2) { v3 = v2; i3 = i2; v2 = v;  i2 = t; }
    else if (v > v3) { v3 = v;  i3 = t; }
  }
  for (int vv = 0; vv < VAL_; ++vv) {
    const float o = sc[i1] * vmat[(b * T_ + i1) * SLOT_ + h * VAL_ + vv] +
                    sc[i2] * vmat[(b * T_ + i2) * SLOT_ + h * VAL_ + vv] +
                    sc[i3] * vmat[(b * T_ + i3) * SLOT_ + h * VAL_ + vv];
    Mi_new[(b * NS_ + n) * SLOT_ + h * VAL_ + vv] = o;
  }
}

// ---------------------------------------------------------------------------
// rel[b,i,d,e] = rel_mem_bias[i,d,e]  (broadcast over b)
// ---------------------------------------------------------------------------
__global__ void rel_init(const float* __restrict__ rel_bias,
                         float* __restrict__ rel) {
  const long total = (long)B_ * NS_ * SLOT_ * SLOT_;
  const long per_b = (long)NS_ * SLOT_ * SLOT_;
  for (long i = blockIdx.x * (long)blockDim.x + threadIdx.x; i < total;
       i += (long)gridDim.x * blockDim.x)
    rel[i] = rel_bias[i % per_b];
}

// ---------------------------------------------------------------------------
// Per-step preamble, one block (256 thr) per batch b:
//   Vtr = Mi_new . rel ; g = rowsum(Vtr) ; h = (Mi_new + g*c2)^T
//   qkv = h @ W_qkv + b ; LayerNorm over (96,24) ; s = tanh(qs*ks) ; vs
// ---------------------------------------------------------------------------
__global__ void step_pre(const float* __restrict__ rel,
                         const float* __restrict__ Mi_new,
                         const float* __restrict__ c2all, int t,
                         const float* __restrict__ W_qkv,
                         const float* __restrict__ b_qkv,
                         const float* __restrict__ ln_g,
                         const float* __restrict__ ln_b,
                         float* __restrict__ s_out,
                         float* __restrict__ vs_out) {
  const int b = blockIdx.x;
  const int tid = threadIdx.x;
  __shared__ float lMi[NS_ * SLOT_];
  __shared__ float lVtr[NS_ * SLOT_];
  __shared__ float lc2[SLOT_];
  __shared__ float lg[NS_];
  __shared__ float lqkv[SLOT_ * 3 * NS_];
  __shared__ float red_a[256];
  __shared__ float red_b[256];

  for (int i = tid; i < NS_ * SLOT_; i += 256) lMi[i] = Mi_new[b * NS_ * SLOT_ + i];
  for (int i = tid; i < SLOT_; i += 256) lc2[i] = c2all[(b * T_ + t) * SLOT_ + i];
  __syncthreads();

  // Vtr[n,f] = sum_d Mi_new[n,d] * rel[b,n,d,f]
  for (int o = tid; o < NS_ * SLOT_; o += 256) {
    const int n = o / SLOT_, f = o % SLOT_;
    const float* rp = rel + ((long)(b * NS_ + n) * SLOT_) * SLOT_ + f;
    float acc = 0.0f;
    for (int d = 0; d < SLOT_; ++d) acc += lMi[n * SLOT_ + d] * rp[(long)d * SLOT_];
    lVtr[o] = acc;
  }
  __syncthreads();
  if (tid < NS_) {
    float s = 0.0f;
    for (int f = 0; f < SLOT_; ++f) s += lVtr[tid * SLOT_ + f];
    lg[tid] = s;
  }
  __syncthreads();

  // qkv[f,c] = sum_n (Mi_new[n,f] + g[n]*c2[f]) * W_qkv[n,c] + b_qkv[c]
  for (int o = tid; o < SLOT_ * 3 * NS_; o += 256) {
    const int f = o / (3 * NS_), c = o % (3 * NS_);
    float acc = b_qkv[c];
#pragma unroll
    for (int n = 0; n < NS_; ++n)
      acc += (lMi[n * SLOT_ + f] + lg[n] * lc2[f]) * W_qkv[n * 3 * NS_ + c];
    lqkv[o] = acc;
  }
  __syncthreads();

  // LayerNorm over all 96*24 elements
  float ps = 0.0f, pq = 0.0f;
  for (int o = tid; o < SLOT_ * 3 * NS_; o += 256) {
    const float v = lqkv[o];
    ps += v;
    pq += v * v;
  }
  red_a[tid] = ps;
  red_b[tid] = pq;
  __syncthreads();
  for (int sh = 128; sh > 0; sh >>= 1) {
    if (tid < sh) { red_a[tid] += red_a[tid + sh]; red_b[tid] += red_b[tid + sh]; }
    __syncthreads();
  }
  const float mu  = red_a[0] / (float)(SLOT_ * 3 * NS_);
  const float var = red_b[0] / (float)(SLOT_ * 3 * NS_) - mu * mu;
  const float inv = rsqrtf(var + EPS_);
  __syncthreads();
  for (int o = tid; o < SLOT_ * 3 * NS_; o += 256)
    lqkv[o] = (lqkv[o] - mu) * inv * ln_g[o] + ln_b[o];
  __syncthreads();

  // s[i,j,d] = tanh(qs[i,d]*ks[j,d]);  qs[i,d]=lqkv[d*24+i], ks[j,d]=lqkv[d*24+8+j]
  for (int o = tid; o < NS_ * NS_ * SLOT_; o += 256) {
    const int i = o / (NS_ * SLOT_);
    const int j = (o / SLOT_) % NS_;
    const int d = o % SLOT_;
    s_out[(long)b * NS_ * NS_ * SLOT_ + o] =
        tanhf(lqkv[d * 3 * NS_ + i] * lqkv[d * 3 * NS_ + NS_ + j]);
  }
  // vs[j,e] = lqkv[e*24 + 16 + j]
  for (int o = tid; o < NS_ * SLOT_; o += 256) {
    const int j = o / SLOT_, e = o % SLOT_;
    vs_out[(long)b * NS_ * SLOT_ + o] = lqkv[e * 3 * NS_ + 2 * NS_ + j];
  }
}

// ---------------------------------------------------------------------------
// rel[b,i,d,e] += alpha * sum_j s[b,i,j,d] * vs[b,j,e]   (rank-8 update)
// One block (256 thr) per (b,i).
// ---------------------------------------------------------------------------
__global__ void rel_update(float* __restrict__ rel,
                           const float* __restrict__ s_in,
                           const float* __restrict__ vs_in,
                           const float* __restrict__ alpha1) {
  const int bi = blockIdx.x;
  const int b = bi >> 3, i = bi & 7;
  const int tid = threadIdx.x;
  __shared__ float lS[NS_ * SLOT_];
  __shared__ float lV[NS_ * SLOT_];
  for (int o = tid; o < NS_ * SLOT_; o += 256) {
    lS[o] = s_in[(long)b * NS_ * NS_ * SLOT_ + i * NS_ * SLOT_ + o];
    lV[o] = vs_in[(long)b * NS_ * SLOT_ + o];
  }
  __syncthreads();
  const float al = alpha1[0];
  float* rp = rel + (long)bi * SLOT_ * SLOT_;
  for (int o = tid; o < SLOT_ * SLOT_; o += 256) {
    const int d = o / SLOT_, e = o % SLOT_;
    float acc = 0.0f;
#pragma unroll
    for (int j = 0; j < NS_; ++j) acc += lS[j * SLOT_ + d] * lV[j * SLOT_ + e];
    rp[o] += al * acc;
  }
}

// ---------------------------------------------------------------------------
extern "C" void kernel_launch(void* const* d_in, const int* in_sizes, int n_in,
                              void* d_out, int out_size, void* d_ws, size_t ws_size,
                              hipStream_t stream) {
  const float* input_step = (const float*)d_in[0];
  const float* Mi     = (const float*)d_in[1];
  const float* W_in   = (const float*)d_in[2];
  const float* b_in   = (const float*)d_in[3];
  const float* Wq     = (const float*)d_in[4];
  const float* bq     = (const float*)d_in[5];
  const float* Wk     = (const float*)d_in[6];
  const float* bk     = (const float*)d_in[7];
  const float* Wv     = (const float*)d_in[8];
  const float* bv     = (const float*)d_in[9];
  const float* W_qkv  = (const float*)d_in[10];
  const float* b_qkv  = (const float*)d_in[11];
  const float* ln_g   = (const float*)d_in[12];
  const float* ln_b   = (const float*)d_in[13];
  const float* alpha1 = (const float*)d_in[14];
  const float* W_in2  = (const float*)d_in[15];
  const float* b_in2  = (const float*)d_in[16];
  const float* W_rel  = (const float*)d_in[17];
  const float* b_rel  = (const float*)d_in[18];
  const float* W_rel3 = (const float*)d_in[19];
  const float* b_rel3 = (const float*)d_in[20];
  const float* rel_bias = (const float*)d_in[21];
  float* out = (float*)d_out;

  // workspace carve-up (floats)
  float* ws = (float*)d_ws;
  float* x_bt  = ws;                         // 4096*96
  float* inp   = x_bt  + 4096 * 96;          // 4096*96
  float* kmat  = inp   + 4096 * 96;          // 4096*128
  float* vmat  = kmat  + 4096 * 128;         // 4096*96
  float* c2all = vmat  + 4096 * 96;          // 4096*96
  float* qmat  = c2all + 4096 * 96;          // 1024*128
  float* Mi_new = qmat + 1024 * 128;         // 1024*96
  float* rel   = Mi_new + 1024 * 96;         // 1024*96*96  (37.7 MB)
  float* s_buf = rel + (long)1024 * 96 * 96; // 128*8*8*96
  float* vs_buf = s_buf + 128 * 8 * 8 * 96;  // 128*8*96
  float* r_vec = vs_buf + 128 * 8 * 96;      // 128*768

  // ---- pre-scan ----
  transpose_in<<<(T_ * B_ * IN_ + 255) / 256, 256, 0, stream>>>(input_step, x_bt);

  // inp = x_bt @ W_in + b_in          (4096 x 96 x 96)
  gemm_bf16_wmma<<<(4096 / 64) * (96 / 32), 256, 0, stream>>>(
      x_bt, W_in, b_in, inp, 4096, 96, 96);
  // kmat = inp @ Wk + bk              (4096 x 128 x 96)
  gemm_bf16_wmma<<<(4096 / 64) * (128 / 32), 256, 0, stream>>>(
      inp, Wk, bk, kmat, 4096, 128, 96);
  // vmat = inp @ Wv + bv              (4096 x 96 x 96)
  gemm_bf16_wmma<<<(4096 / 64) * (96 / 32), 256, 0, stream>>>(
      inp, Wv, bv, vmat, 4096, 96, 96);
  // c2all = inp @ W_in2 + b_in2       (4096 x 96 x 96)
  gemm_bf16_wmma<<<(4096 / 64) * (96 / 32), 256, 0, stream>>>(
      inp, W_in2, b_in2, c2all, 4096, 96, 96);
  // qmat = Mi @ Wq + bq               (1024 x 128 x 96)
  gemm_bf16_wmma<<<(1024 / 64) * (128 / 32), 256, 0, stream>>>(
      Mi, Wq, bq, qmat, 1024, 128, 96);

  attention_topk<<<(B_ * H_ * NS_ + 255) / 256, 256, 0, stream>>>(
      qmat, kmat, vmat, Mi_new);

  rel_init<<<4096, 256, 0, stream>>>(rel_bias, rel);

  // ---- sequential scan over T steps ----
  for (int t = 0; t < T_; ++t) {
    step_pre<<<B_, 256, 0, stream>>>(rel, Mi_new, c2all, t, W_qkv, b_qkv,
                                     ln_g, ln_b, s_buf, vs_buf);
    rel_update<<<B_ * NS_, 256, 0, stream>>>(rel, s_buf, vs_buf, alpha1);
    // r_vec = rel.reshape(1024,9216) @ W_rel + b_rel   (dominant GEMM)
    gemm_bf16_wmma<<<(1024 / 64) * (96 / 32), 256, 0, stream>>>(
        rel, W_rel, b_rel, r_vec, 1024, 96, 9216);
    // out[t] = r_vec.reshape(128,768) @ W_rel3 + b_rel3
    gemm_bf16_wmma<<<(128 / 64) * (64 / 32), 256, 0, stream>>>(
        r_vec, W_rel3, b_rel3, out + (long)t * B_ * OUTA_, 128, 64, 768);
  }
}
